// ScaledDotProductAttention_51969104282201
// MI455X (gfx1250) — matrix operand order repaired and checked by reading, not compiled, not run on hardware
//
#include <hip/hip_runtime.h>

typedef __attribute__((ext_vector_type(16))) _Float16 v16h;
typedef __attribute__((ext_vector_type(8)))  _Float16 v8h;
typedef __attribute__((ext_vector_type(8)))  float    v8f;

#define ATTN_B 32
#define ATTN_S 2048
#define ATTN_D 64

#define KH_BYTES  ((size_t)ATTN_B * ATTN_S * ATTN_D * 2)   // 8 MiB
#define VT_BYTES  ((size_t)ATTN_B * ATTN_S * ATTN_D * 2)   // 8 MiB
#define WS_NEEDED (KH_BYTES + VT_BYTES)

// softmax scale (1/sqrt(64)) folded together with log2(e) so the per-score
// work is a single v_exp_f32 (exp2): exp(scale*s) == exp2((scale*log2e)*s).
#define QSCALE (0.125f * 1.4426950408889634f)

// exp2 lowering straight to v_exp_f32
#define EXP2F(x) __builtin_amdgcn_exp2f(x)

// ---------------------------------------------------------------------------
// Pre-pass 1: K (f32, [b][key][dim]) -> Kh (f16, same layout).
// ---------------------------------------------------------------------------
__global__ __launch_bounds__(256)
void convert_k_f16(const float* __restrict__ K, _Float16* __restrict__ Kh)
{
    const size_t i = ((size_t)blockIdx.x * blockDim.x + threadIdx.x) * 8;
    const float4* s = reinterpret_cast<const float4*>(K + i);
    float4 a = s[0], b = s[1];
    v8h o;
    o[0] = (_Float16)a.x; o[1] = (_Float16)a.y; o[2] = (_Float16)a.z; o[3] = (_Float16)a.w;
    o[4] = (_Float16)b.x; o[5] = (_Float16)b.y; o[6] = (_Float16)b.z; o[7] = (_Float16)b.w;
    *reinterpret_cast<v8h*>(Kh + i) = o;
}

// ---------------------------------------------------------------------------
// Pre-pass 2: V (f32, [b][key][dim]) -> VT (f16, [b][dim][key]).
// ---------------------------------------------------------------------------
__global__ __launch_bounds__(256)
void transpose_v_f16(const float* __restrict__ V, _Float16* __restrict__ VT)
{
    __shared__ _Float16 tile[64][72];
    const int b  = blockIdx.x >> 5;
    const int k0 = (blockIdx.x & 31) << 6;
    const int t  = threadIdx.x;
    const int row = t >> 2;
    const int c0  = (t & 3) << 4;

    const float4* src = reinterpret_cast<const float4*>(
        V + ((size_t)b * ATTN_S + k0 + row) * ATTN_D + c0);
    #pragma unroll
    for (int i = 0; i < 4; ++i) {
        float4 x = src[i];
        tile[row][c0 + 4*i + 0] = (_Float16)x.x;
        tile[row][c0 + 4*i + 1] = (_Float16)x.y;
        tile[row][c0 + 4*i + 2] = (_Float16)x.z;
        tile[row][c0 + 4*i + 3] = (_Float16)x.w;
    }
    __syncthreads();

    const int dim = t >> 2;
    const int kk  = (t & 3) << 4;
    _Float16* dst = VT + ((size_t)b * ATTN_D + dim) * ATTN_S + k0 + kk;
    v8h o0, o1;
    #pragma unroll
    for (int j = 0; j < 8; ++j) {
        o0[j] = tile[kk + j][dim];
        o1[j] = tile[kk + 8 + j][dim];
    }
    reinterpret_cast<v8h*>(dst)[0] = o0;
    reinterpret_cast<v8h*>(dst)[1] = o1;
}

// ---------------------------------------------------------------------------
// One 32-key block: 4 WMMAs for S^T = K*Q^T, exp2 softmax weights (scale
// pre-folded into Q), 4 WMMAs for acc += P*V. MASKED=false for blocks fully
// below the diagonal (no compares/cndmask at all).
// ---------------------------------------------------------------------------
template<bool MASKED>
__device__ __forceinline__ void attn_block(
    const _Float16* __restrict__ kbase, const _Float16* __restrict__ vtbase,
    const v16h& qB0, const v16h& qB1, v8f acc[4], float& lsum,
    int kb, int qidx, int lg, int ll)
{
    // K rows in A-layout: M = key = ll, halfs = dims; pure b128 loads.
    const _Float16* kr0 = kbase + (size_t)kb * ATTN_D;   // key kb+ll
    const _Float16* kr1 = kr0 + 16 * ATTN_D;             // key kb+16+ll
    v8h k00 = *reinterpret_cast<const v8h*>(kr0);
    v8h k01 = *reinterpret_cast<const v8h*>(kr0 + 16);
    v8h k02 = *reinterpret_cast<const v8h*>(kr0 + 32);
    v8h k03 = *reinterpret_cast<const v8h*>(kr0 + 48);
    v8h k10 = *reinterpret_cast<const v8h*>(kr1);
    v8h k11 = *reinterpret_cast<const v8h*>(kr1 + 16);
    v8h k12 = *reinterpret_cast<const v8h*>(kr1 + 32);
    v8h k13 = *reinterpret_cast<const v8h*>(kr1 + 48);
    v16h a00 = __builtin_shufflevector(k00, k01, 0,1,2,3,4,5,6,7,8,9,10,11,12,13,14,15);
    v16h a01 = __builtin_shufflevector(k02, k03, 0,1,2,3,4,5,6,7,8,9,10,11,12,13,14,15);
    v16h a10 = __builtin_shufflevector(k10, k11, 0,1,2,3,4,5,6,7,8,9,10,11,12,13,14,15);
    v16h a11 = __builtin_shufflevector(k12, k13, 0,1,2,3,4,5,6,7,8,9,10,11,12,13,14,15);

    // S^T = K * Q^T (C-layout: lane -> query = ll, VGPR r -> key kb+8lg+r (+16)).
    v8f s0 = (v8f){0.f,0.f,0.f,0.f,0.f,0.f,0.f,0.f};
    v8f s1 = s0;
    s0 = __builtin_amdgcn_wmma_f32_16x16x32_f16(false, a00, false, qB0, (short)0, s0, false, false);
    s0 = __builtin_amdgcn_wmma_f32_16x16x32_f16(false, a01, false, qB1, (short)0, s0, false, false);
    s1 = __builtin_amdgcn_wmma_f32_16x16x32_f16(false, a10, false, qB0, (short)0, s1, false, false);
    s1 = __builtin_amdgcn_wmma_f32_16x16x32_f16(false, a11, false, qB1, (short)0, s1, false, false);

    // exp2 + (optional) causal mask + row-sum; P built directly in A-layout.
    v16h pA;
    const int keyb = kb + 8 * lg;
    #pragma unroll
    for (int r = 0; r < 8; ++r) {
        float e0 = EXP2F(s0[r]);
        float e1 = EXP2F(s1[r]);
        if (MASKED) {
            if (keyb + r      > qidx) e0 = 0.f;
            if (keyb + r + 16 > qidx) e1 = 0.f;
        }
        lsum += e0 + e1;
        pA[r]     = (_Float16)e0;
        pA[r + 8] = (_Float16)e1;
    }

    // acc += P(16x32) * V(32x64); B-operand = keys 8lg..8lg+7 and 16+8lg..16+8lg+7 of V^T row.
    #pragma unroll
    for (int d = 0; d < 4; ++d) {
        const _Float16* vt = vtbase + (size_t)(d * 16 + ll) * ATTN_S + kb;
        v16h vb = __builtin_shufflevector(*reinterpret_cast<const v8h*>(vt), *reinterpret_cast<const v8h*>(vt + 16), 0,1,2,3,4,5,6,7,8,9,10,11,12,13,14,15);
        acc[d] = __builtin_amdgcn_wmma_f32_16x16x32_f16(false, pA, false, vb, (short)0, acc[d], false, false);
    }
}

// ---------------------------------------------------------------------------
// Main kernel: one wave = 16 queries x D=64 of one batch; streams causal
// 32-key blocks. Full blocks take the mask-free path; exactly one diagonal
// block per q-tile is masked. Reference numerics preserved: plain exp
// softmax (via exp2 with folded scale), mask after exp, normalize once.
// ---------------------------------------------------------------------------
__global__ __launch_bounds__(256)
void sdpa_causal_wmma_f16src(const float* __restrict__ Q,
                             const _Float16* __restrict__ Kh,
                             const _Float16* __restrict__ VT,
                             float* __restrict__ O)
{
    const int lane = threadIdx.x & 31;
    const int wave = threadIdx.x >> 5;
    const int tile = blockIdx.x * (blockDim.x >> 5) + wave;
    const int b    = tile >> 7;
    const int q0   = (tile & 127) << 4;

    const int lg = lane >> 4;
    const int ll = lane & 15;
    const size_t bb = (size_t)b * ATTN_S * ATTN_D;

    // Q^T in B-layout with scale*log2e folded in (converted once per wave).
    const float* qrow = Q + bb + (size_t)(q0 + ll) * ATTN_D + 8 * lg;
    v16h qB0, qB1;
    #pragma unroll
    for (int h = 0; h < 16; ++h) {
        qB0[h] = (_Float16)(qrow[h + (h >> 3) * 8] * QSCALE);
        qB1[h] = (_Float16)(qrow[32 + h + (h >> 3) * 8] * QSCALE);
    }

    v8f acc[4];
    #pragma unroll
    for (int d = 0; d < 4; ++d)
        acc[d] = (v8f){0.f, 0.f, 0.f, 0.f, 0.f, 0.f, 0.f, 0.f};
    float lsum = 0.f;

    const int qidx = q0 + ll;
    const _Float16* kbase  = Kh + bb + (size_t)ll * ATTN_D + 8 * lg;
    const _Float16* vtbase = VT + (size_t)b * ATTN_D * ATTN_S + 8 * lg;

    const int kb_edge = q0 & ~31;      // the single partially-masked block
    for (int kb = 0; kb < kb_edge; kb += 32) {
        // Unconditional speculative prefetch of the next block (stays inside ws).
        __builtin_prefetch(kbase + (size_t)(kb + 32) * ATTN_D, 0, 0);
        __builtin_prefetch(vtbase + (size_t)lane * ATTN_S + kb + 32, 0, 0);
        attn_block<false>(kbase, vtbase, qB0, qB1, acc, lsum, kb, qidx, lg, ll);
    }
    attn_block<true>(kbase, vtbase, qB0, qB1, acc, lsum, kb_edge, qidx, lg, ll);

    lsum += __shfl_xor(lsum, 16, 32);

    float* obase = O + bb + (size_t)(q0 + 8 * lg) * ATTN_D + ll;
    #pragma unroll
    for (int r = 0; r < 8; ++r) {
        const float denom = __shfl(lsum, r + 8 * lg, 32);
        const float inv = 1.0f / denom;
        #pragma unroll
        for (int d = 0; d < 4; ++d)
            obase[(size_t)r * ATTN_D + d * 16] = acc[d][r] * inv;
    }
}

// ---------------------------------------------------------------------------
// Fallback (ws too small): all-f32-source version, same math.
// ---------------------------------------------------------------------------
__global__ __launch_bounds__(256)
void sdpa_causal_wmma_f32src(const float* __restrict__ Q,
                             const float* __restrict__ K,
                             const float* __restrict__ V,
                             float* __restrict__ O)
{
    const int lane = threadIdx.x & 31;
    const int wave = threadIdx.x >> 5;
    const int tile = blockIdx.x * (blockDim.x >> 5) + wave;
    const int b    = tile >> 7;
    const int q0   = (tile & 127) << 4;

    const int lg = lane >> 4;
    const int ll = lane & 15;
    const size_t bb = (size_t)b * ATTN_S * ATTN_D;

    const float* qrow = Q + bb + (size_t)(q0 + ll) * ATTN_D + 8 * lg;
    v16h qB0, qB1;
    #pragma unroll
    for (int h = 0; h < 16; ++h) {
        qB0[h] = (_Float16)(qrow[h + (h >> 3) * 8] * QSCALE);
        qB1[h] = (_Float16)(qrow[32 + h + (h >> 3) * 8] * QSCALE);
    }

    v8f acc[4];
    #pragma unroll
    for (int d = 0; d < 4; ++d)
        acc[d] = (v8f){0.f, 0.f, 0.f, 0.f, 0.f, 0.f, 0.f, 0.f};
    float lsum = 0.f;
    const int qidx = q0 + ll;
    const int qhi  = q0 + 15;

    for (int kb = 0; kb <= qhi; kb += 32) {
        const float* kr0 = K + bb + (size_t)(kb + ll) * ATTN_D + 8 * lg;
        const float* kr1 = kr0 + 16 * ATTN_D;
        v16h a00, a01, a10, a11;
        #pragma unroll
        for (int h = 0; h < 8; ++h) {
            a00[h] = (_Float16)kr0[h];      a00[h+8] = (_Float16)kr0[h + 16];
            a01[h] = (_Float16)kr0[h + 32]; a01[h+8] = (_Float16)kr0[h + 48];
            a10[h] = (_Float16)kr1[h];      a10[h+8] = (_Float16)kr1[h + 16];
            a11[h] = (_Float16)kr1[h + 32]; a11[h+8] = (_Float16)kr1[h + 48];
        }
        v8f s0 = (v8f){0.f,0.f,0.f,0.f,0.f,0.f,0.f,0.f};
        v8f s1 = s0;
        s0 = __builtin_amdgcn_wmma_f32_16x16x32_f16(false, a00, false, qB0, (short)0, s0, false, false);
        s0 = __builtin_amdgcn_wmma_f32_16x16x32_f16(false, a01, false, qB1, (short)0, s0, false, false);
        s1 = __builtin_amdgcn_wmma_f32_16x16x32_f16(false, a10, false, qB0, (short)0, s1, false, false);
        s1 = __builtin_amdgcn_wmma_f32_16x16x32_f16(false, a11, false, qB1, (short)0, s1, false, false);

        v16h pA;
        const int keyb = kb + 8 * lg;
        #pragma unroll
        for (int r = 0; r < 8; ++r) {
            float e0 = (keyb + r      <= qidx) ? EXP2F(s0[r]) : 0.f;
            float e1 = (keyb + r + 16 <= qidx) ? EXP2F(s1[r]) : 0.f;
            lsum += e0 + e1;
            pA[r] = (_Float16)e0;
            pA[r + 8] = (_Float16)e1;
        }

        const float* vb_base = V + bb + (size_t)(kb + 8 * lg) * ATTN_D + ll;
        #pragma unroll
        for (int d = 0; d < 4; ++d) {
            v16h vb;
            #pragma unroll
            for (int h = 0; h < 16; ++h)
                vb[h] = (_Float16)vb_base[(size_t)(h + (h >> 3) * 8) * ATTN_D + d * 16];
            acc[d] = __builtin_amdgcn_wmma_f32_16x16x32_f16(false, pA, false, vb, (short)0, acc[d], false, false);
        }
    }

    lsum += __shfl_xor(lsum, 16, 32);
    float* obase = O + bb + (size_t)(q0 + 8 * lg) * ATTN_D + ll;
    #pragma unroll
    for (int r = 0; r < 8; ++r) {
        const float denom = __shfl(lsum, r + 8 * lg, 32);
        const float inv = 1.0f / denom;
        #pragma unroll
        for (int d = 0; d < 4; ++d)
            obase[(size_t)r * ATTN_D + d * 16] = acc[d][r] * inv;
    }
}

extern "C" void kernel_launch(void* const* d_in, const int* in_sizes, int n_in,
                              void* d_out, int out_size, void* d_ws, size_t ws_size,
                              hipStream_t stream) {
    (void)in_sizes; (void)n_in; (void)out_size;
    const float* Q = (const float*)d_in[0];
    const float* K = (const float*)d_in[1];
    const float* V = (const float*)d_in[2];
    // d_in[3] is the causal mask; reproduced analytically in-kernel.
    float* O = (float*)d_out;

    const int waves_per_block = 8;
    const int tiles = ATTN_B * (ATTN_S / 16);       // 4096 q-tiles
    dim3 grid(tiles / waves_per_block);             // 512 blocks

    if (d_ws != nullptr && ws_size >= WS_NEEDED) {
        _Float16* Kh = (_Float16*)d_ws;
        _Float16* VT = (_Float16*)((char*)d_ws + KH_BYTES);

        const size_t n_elems = (size_t)ATTN_B * ATTN_S * ATTN_D;
        convert_k_f16<<<dim3((unsigned)(n_elems / 8 / 256)), 256, 0, stream>>>(K, Kh);
        transpose_v_f16<<<dim3(ATTN_B * (ATTN_S / 64)), 256, 0, stream>>>(V, VT);
        sdpa_causal_wmma_f16src<<<grid, waves_per_block * 32, 0, stream>>>(Q, Kh, VT, O);
    } else {
        sdpa_causal_wmma_f32src<<<grid, waves_per_block * 32, 0, stream>>>(Q, K, V, O);
    }
}
